// DeformableTransformerDecoderLayer_23356032156018
// MI455X (gfx1250) — matrix operand order repaired and checked by reading, not compiled, not run on hardware
//
#include <hip/hip_runtime.h>
#include <cstdint>

// ---------------------------------------------------------------------------
// DeformableTransformerDecoderLayer for MI455X (gfx1250, wave32, WMMA bf16)
// D_MODEL=256, H=8, dh=32, L=4, P=4, D_FFN=1024, BS=4, NQ=1000, LEN_MEM=21760
// ---------------------------------------------------------------------------

typedef __bf16 bf16;
typedef __attribute__((ext_vector_type(16))) __bf16 v16bf;
typedef __attribute__((ext_vector_type(8)))  float  v8f;

union U32x8 { uint4 q[2]; v16bf v; };

static __device__ inline v16bf ld_frag(const bf16* p0, const bf16* p1) {
    U32x8 u;
    u.q[0] = *(const uint4*)p0;
    u.q[1] = *(const uint4*)p1;
    return u.v;
}

static __device__ inline v8f v8f_zero() {
    v8f z = {0.f,0.f,0.f,0.f,0.f,0.f,0.f,0.f};
    return z;
}

static __device__ inline v8f wmma_bf16(v16bf a, v16bf b, v8f c) {
    // D = A(16x32) * B(32x16) + C, f32 accumulate
    return __builtin_amdgcn_wmma_f32_16x16x32_bf16(false, a, false, b, (short)0, c, false, false);
}

// ---------------------------------------------------------------------------
// f32 -> bf16 weight conversion
// ---------------------------------------------------------------------------
__global__ void cvt_bf16_kernel(const float* __restrict__ s, bf16* __restrict__ d, int n) {
    int i = blockIdx.x * 256 + threadIdx.x;
    if (i < n) d[i] = (bf16)s[i];
}
__global__ void copy_f32_kernel(const float* __restrict__ s, float* __restrict__ d, int n) {
    int i = blockIdx.x * 256 + threadIdx.x;
    if (i < n) d[i] = s[i];
}

// ---------------------------------------------------------------------------
// Generic tiled GEMM: out[M x Ntot] (+bias,+resid,ReLU) = A[M x K] * W^T
//   W stored [N][K] row-major in bf16 (matches x @ W.T).
//   Block: 128 threads (4 waves). Block tile: 16 rows x 256 cols.
//   QKV fusion: columns >= colSplit use (A1,A1add) instead of (A0,A0add);
//   A_eff = A + Aadd (elementwise) converted to bf16 in LDS.
// ---------------------------------------------------------------------------
template<int K>
__global__ __launch_bounds__(128)
void gemm_kernel(const float* __restrict__ A0, const float* __restrict__ A0add,
                 const float* __restrict__ A1, const float* __restrict__ A1add,
                 int colSplit,
                 const bf16* __restrict__ W, const float* __restrict__ bias,
                 const float* __restrict__ resid, int relu,
                 float* __restrict__ out, int Ntot)
{
    __shared__ __align__(16) bf16 sA[16 * K];
    const int m0 = blockIdx.x * 16;
    const int colBase = blockIdx.y * 256;
    const int tid = threadIdx.x;

    const float* A    = (colBase >= colSplit) ? A1    : A0;
    const float* Aadd = (colBase >= colSplit) ? A1add : A0add;

    for (int i = tid; i < 16 * K; i += 128) {
        int r = i / K, c = i - r * K;
        size_t gi = (size_t)(m0 + r) * K + c;
        float v = A[gi];
        if (Aadd) v += Aadd[gi];
        sA[r * K + c] = (bf16)v;
    }
    __syncthreads();

    const int wave = tid >> 5;
    const int lane = tid & 31;
    const int row  = lane & 15;   // A row / B col / C col
    const int hi   = lane >> 4;

    v8f acc[4];
    #pragma unroll
    for (int j = 0; j < 4; ++j) acc[j] = v8f_zero();

    for (int k = 0; k < K; k += 32) {
        const bf16* ap = sA + row * K + k + hi * 8;
        v16bf a = ld_frag(ap, ap + 16);
        #pragma unroll
        for (int j = 0; j < 4; ++j) {
            int n = colBase + (wave * 4 + j) * 16 + row;
            const bf16* bp = W + (size_t)n * K + k + hi * 8;
            v16bf b = ld_frag(bp, bp + 16);
            acc[j] = wmma_bf16(a, b, acc[j]);
        }
    }

    #pragma unroll
    for (int j = 0; j < 4; ++j) {
        int ncol = colBase + (wave * 4 + j) * 16 + row;
        float bval = bias[ncol];
        #pragma unroll
        for (int i = 0; i < 8; ++i) {
            int m = m0 + i + 8 * hi;
            float v = acc[j][i] + bval;
            if (resid) v += resid[(size_t)m * Ntot + ncol];
            if (relu)  v = v > 0.f ? v : 0.f;
            out[(size_t)m * Ntot + ncol] = v;
        }
    }
}

// ---------------------------------------------------------------------------
// Pack per-(b,h) attention operands, padded to 1024 rows, bf16.
//  Qb,Kb: [32][1024][32]   (Q pre-scaled by 1/sqrt(dh))
//  Vt:    [32][32][1024]   (transposed so B-fragments load contiguously)
// ---------------------------------------------------------------------------
__global__ __launch_bounds__(256)
void pack_attn_kernel(const float* __restrict__ qkv,
                      bf16* __restrict__ Qb, bf16* __restrict__ Kb, bf16* __restrict__ Vt)
{
    const int bh = blockIdx.x, b = bh >> 3, h = bh & 7;
    const float scale = 0.17677669529663687f; // 1/sqrt(32)
    const size_t o = (size_t)bh * 32768;
    for (int i = threadIdx.x; i < 1024 * 32; i += 256) {
        int row = i >> 5, d = i & 31;
        float qv = 0.f, kv = 0.f, vv = 0.f;
        if (row < 1000) {
            size_t base = ((size_t)row * 4 + b) * 768 + h * 32 + d;
            qv = qkv[base] * scale;
            kv = qkv[base + 256];
            vv = qkv[base + 512];
        }
        Qb[o + i] = (bf16)qv;
        Kb[o + i] = (bf16)kv;
        Vt[o + (size_t)d * 1024 + row] = (bf16)vv;
    }
}

// ---------------------------------------------------------------------------
// Self-attention core: one block per (q-tile of 16, (b,h)).
//  S(16x1024) = Qtile * K^T (WMMA) -> masked softmax -> P bf16 in LDS
//  out(16x32) = P * V (WMMA, K split over 4 waves, LDS reduce)
// ---------------------------------------------------------------------------
__global__ __launch_bounds__(128)
void attn_kernel(const bf16* __restrict__ Qb, const bf16* __restrict__ Kb,
                 const bf16* __restrict__ Vt, float* __restrict__ sa)
{
    __shared__ __align__(16) float sS[16 * 1024];   // scores / exp / partials
    __shared__ __align__(16) bf16  sP[16 * 1024];   // normalized probs
    __shared__ float sRed[16 * 8];

    const int q0 = blockIdx.x * 16;
    const int bh = blockIdx.y;
    const int b = bh >> 3, h = bh & 7;
    const int tid = threadIdx.x, wave = tid >> 5, lane = tid & 31;
    const int row = lane & 15, hi = lane >> 4;

    const bf16* Q  = Qb + (size_t)bh * 32768;
    const bf16* Km = Kb + (size_t)bh * 32768;
    const bf16* V  = Vt + (size_t)bh * 32768;

    // ---- scores: each wave does 16 column-tiles of 16 (K-dim = dh = 32) ----
    const bf16* qp = Q + (q0 + row) * 32 + hi * 8;
    v16bf afr = ld_frag(qp, qp + 16);
    for (int j = 0; j < 16; ++j) {
        int ct = wave * 16 + j;
        const bf16* bp = Km + (size_t)(ct * 16 + row) * 32 + hi * 8;
        v16bf bfr = ld_frag(bp, bp + 16);
        v8f s = wmma_bf16(afr, bfr, v8f_zero());
        #pragma unroll
        for (int i = 0; i < 8; ++i)
            sS[(i + 8 * hi) * 1024 + ct * 16 + row] = s[i];
    }
    __syncthreads();

    // ---- masked softmax: 8 threads per row, 128 cols each ----
    {
        const int r = tid >> 3, sub = tid & 7, base = sub * 128;
        float* Srow = sS + r * 1024 + base;
        float mx = -1e30f;
        for (int c = 0; c < 128; ++c) {
            float v = (base + c < 1000) ? Srow[c] : -1e30f;
            mx = fmaxf(mx, v);
        }
        sRed[r * 8 + sub] = mx;
        __syncthreads();
        float rmx = -1e30f;
        for (int i = 0; i < 8; ++i) rmx = fmaxf(rmx, sRed[r * 8 + i]);
        __syncthreads();
        float sum = 0.f;
        for (int c = 0; c < 128; ++c) {
            float e = (base + c < 1000) ? __expf(Srow[c] - rmx) : 0.f;
            Srow[c] = e;
            sum += e;
        }
        sRed[r * 8 + sub] = sum;
        __syncthreads();
        float rs = 0.f;
        for (int i = 0; i < 8; ++i) rs += sRed[r * 8 + i];
        float inv = 1.f / rs;
        bf16* Prow = sP + r * 1024 + base;
        for (int c = 0; c < 128; ++c) Prow[c] = (bf16)(Srow[c] * inv);
    }
    __syncthreads();

    // ---- P @ V: wave w covers K in [w*256, w*256+256), 2 N-tiles (dh=32) ----
    v8f acc0 = v8f_zero(), acc1 = v8f_zero();
    for (int k = wave * 256; k < wave * 256 + 256; k += 32) {
        const bf16* ap = sP + row * 1024 + k + hi * 8;
        v16bf a = ld_frag(ap, ap + 16);
        const bf16* b0 = V + (size_t)(row)      * 1024 + k + hi * 8;
        const bf16* b1 = V + (size_t)(16 + row) * 1024 + k + hi * 8;
        acc0 = wmma_bf16(a, ld_frag(b0, b0 + 16), acc0);
        acc1 = wmma_bf16(a, ld_frag(b1, b1 + 16), acc1);
    }
    // partial reduce across 4 waves via LDS (reuse sS)
    #pragma unroll
    for (int i = 0; i < 8; ++i) {
        int m = i + 8 * hi;
        sS[wave * 512 +       m * 16 + row] = acc0[i];
        sS[wave * 512 + 256 + m * 16 + row] = acc1[i];
    }
    __syncthreads();
    for (int e = tid; e < 512; e += 128) {
        int m = e >> 5, n = e & 31;
        float s = 0.f;
        for (int w = 0; w < 4; ++w)
            s += sS[w * 512 + (n >> 4) * 256 + m * 16 + (n & 15)];
        int q = q0 + m;
        if (q < 1000)
            sa[((size_t)q * 4 + b) * 256 + h * 32 + n] = s;
    }
}

// ---------------------------------------------------------------------------
// Row-wise LayerNorm over 256 channels.
// ---------------------------------------------------------------------------
__global__ __launch_bounds__(256)
void ln_kernel(const float* __restrict__ x, const float* __restrict__ g,
               const float* __restrict__ bta, float* __restrict__ out)
{
    __shared__ float red[256];
    const int r = blockIdx.x, t = threadIdx.x;
    float v = x[(size_t)r * 256 + t];
    red[t] = v; __syncthreads();
    for (int s = 128; s > 0; s >>= 1) { if (t < s) red[t] += red[t + s]; __syncthreads(); }
    float mean = red[0] * (1.f / 256.f);
    __syncthreads();
    float d = v - mean;
    red[t] = d * d; __syncthreads();
    for (int s = 128; s > 0; s >>= 1) { if (t < s) red[t] += red[t + s]; __syncthreads(); }
    float var = red[0] * (1.f / 256.f);
    out[(size_t)r * 256 + t] = d * rsqrtf(var + 1e-5f) * g[t] + bta[t];
}

// ---------------------------------------------------------------------------
// Sampling-offset / attention-weight postprocess: per row r=(q*4+b),
// thread t = h*16 + (l*4+p). Softmax over 16 (per head) via lane shuffles,
// loc = ref_xy + t * (ref_zw - ref_xy).
// ---------------------------------------------------------------------------
__global__ __launch_bounds__(128)
void offaw_post_kernel(const float* __restrict__ offaw, const float* __restrict__ ref,
                       float* __restrict__ loc, float* __restrict__ awn)
{
    const int r = blockIdx.x, t = threadIdx.x;
    const int lp = t & 15, l = lp >> 2;
    float tv = offaw[(size_t)r * 256 + t];
    float av = offaw[(size_t)r * 256 + 128 + t];
    float m = av;
    #pragma unroll
    for (int o = 8; o >= 1; o >>= 1) m = fmaxf(m, __shfl_xor(m, o));
    float e = __expf(av - m);
    float s = e;
    #pragma unroll
    for (int o = 8; o >= 1; o >>= 1) s += __shfl_xor(s, o);
    awn[(size_t)r * 128 + t] = e / s;
    const float* rf = ref + ((size_t)r * 4 + l) * 4;
    loc[((size_t)r * 128 + t) * 2 + 0] = rf[0] + tv * (rf[2] - rf[0]);
    loc[((size_t)r * 128 + t) * 2 + 1] = rf[1] + tv * (rf[3] - rf[1]);
}

// ---------------------------------------------------------------------------
// Multi-scale deformable bilinear gather: block per row r, wave per head,
// lane per channel. 16 (l,p) points x 4 taps; 32 contiguous f32 per tap/wave.
// ---------------------------------------------------------------------------
__global__ __launch_bounds__(256)
void msdeform_kernel(const float* __restrict__ value, const float* __restrict__ loc,
                     const float* __restrict__ awn, float* __restrict__ msd)
{
    const int r = blockIdx.x, b = r & 3;
    const int tid = threadIdx.x, h = tid >> 5, d = tid & 31;
    const int spat[4]  = {128, 64, 32, 16};
    const int loff[4]  = {0, 16384, 20480, 21504};
    const float* lc = loc + (size_t)r * 256 + h * 32;   // [16][2]
    const float* aw = awn + (size_t)r * 128 + h * 16;
    float acc = 0.f;
    #pragma unroll
    for (int l = 0; l < 4; ++l) {
        const int Hs = spat[l], Ws = spat[l];
        const float* vbase = value + ((size_t)b * 21760 + loff[l]) * 256 + h * 32 + d;
        #pragma unroll
        for (int p = 0; p < 4; ++p) {
            int lp = l * 4 + p;
            float lx = lc[lp * 2 + 0] * Ws - 0.5f;
            float ly = lc[lp * 2 + 1] * Hs - 0.5f;
            float x0f = floorf(lx), y0f = floorf(ly);
            float fx = lx - x0f, fy = ly - y0f;
            int x0 = (int)x0f, y0 = (int)y0f;
            float s = 0.f;
            #pragma unroll
            for (int dy = 0; dy < 2; ++dy)
                #pragma unroll
                for (int dx = 0; dx < 2; ++dx) {
                    int xi = x0 + dx, yi = y0 + dy;
                    if (xi >= 0 && xi < Ws && yi >= 0 && yi < Hs) {
                        float w = (dx ? fx : 1.f - fx) * (dy ? fy : 1.f - fy);
                        s += w * vbase[((size_t)yi * Ws + xi) * 256];
                    }
                }
            acc += aw[lp] * s;
        }
    }
    msd[(size_t)r * 256 + h * 32 + d] = acc;
}

// ---------------------------------------------------------------------------
// Host launch
// ---------------------------------------------------------------------------
extern "C" void kernel_launch(void* const* d_in, const int* in_sizes, int n_in,
                              void* d_out, int out_size, void* d_ws, size_t ws_size,
                              hipStream_t stream) {
    (void)in_sizes; (void)n_in; (void)out_size; (void)ws_size;
    const float* tgt      = (const float*)d_in[0];
    const float* pos      = (const float*)d_in[1];
    const float* ref      = (const float*)d_in[2];
    const float* memory   = (const float*)d_in[3];
    // d_in[4] = memory_spatial_shapes (int64) — hardcoded
    const float* sa_in_w  = (const float*)d_in[5];
    const float* sa_in_b  = (const float*)d_in[6];
    const float* sa_out_w = (const float*)d_in[7];
    const float* sa_out_b = (const float*)d_in[8];
    const float* off_w    = (const float*)d_in[9];
    const float* off_b    = (const float*)d_in[10];
    const float* aw_w     = (const float*)d_in[11];
    const float* aw_b     = (const float*)d_in[12];
    const float* val_w    = (const float*)d_in[13];
    const float* val_b    = (const float*)d_in[14];
    const float* proj_w   = (const float*)d_in[15];
    const float* proj_b   = (const float*)d_in[16];
    const float* lin1_w   = (const float*)d_in[17];
    const float* lin1_b   = (const float*)d_in[18];
    const float* lin2_w   = (const float*)d_in[19];
    const float* lin2_b   = (const float*)d_in[20];
    const float* ln1_g    = (const float*)d_in[21];
    const float* ln1_bb   = (const float*)d_in[22];
    const float* ln2_g    = (const float*)d_in[23];
    const float* ln2_bb   = (const float*)d_in[24];
    const float* ln3_g    = (const float*)d_in[25];
    const float* ln3_bb   = (const float*)d_in[26];

    const int M = 4000;             // NQ*BS rows
    const int MV = 4 * 21760;       // value rows
    char* base = (char*)d_ws;
    size_t cur = 0;
    auto alloc = [&](size_t bytes) -> char* {
        char* p = base + cur;
        cur = (cur + bytes + 255) & ~(size_t)255;
        return p;
    };
    bf16* W_SAIN  = (bf16*)alloc((size_t)768 * 256 * 2);
    bf16* W_SAOUT = (bf16*)alloc((size_t)256 * 256 * 2);
    bf16* W_OFFAW = (bf16*)alloc((size_t)256 * 256 * 2);
    bf16* W_VAL   = (bf16*)alloc((size_t)256 * 256 * 2);
    bf16* W_PROJ  = (bf16*)alloc((size_t)256 * 256 * 2);
    bf16* W_LIN1  = (bf16*)alloc((size_t)1024 * 256 * 2);
    bf16* W_LIN2  = (bf16*)alloc((size_t)256 * 1024 * 2);
    float* B_OFFAW = (float*)alloc(256 * 4);
    float* QKV    = (float*)alloc((size_t)M * 768 * 4);
    bf16*  QB     = (bf16*)alloc((size_t)32 * 32768 * 2);
    bf16*  KB     = (bf16*)alloc((size_t)32 * 32768 * 2);
    bf16*  VT     = (bf16*)alloc((size_t)32 * 32768 * 2);
    float* SA     = (float*)alloc((size_t)M * 256 * 4);
    float* Y      = (float*)alloc((size_t)M * 256 * 4);
    float* TGT2   = (float*)alloc((size_t)M * 256 * 4);
    float* OFFAW  = (float*)alloc((size_t)M * 256 * 4);
    float* LOC    = (float*)alloc((size_t)M * 256 * 4);
    float* AWN    = (float*)alloc((size_t)M * 128 * 4);
    float* MSD    = (float*)alloc((size_t)M * 256 * 4);
    float* TGT3   = (float*)alloc((size_t)M * 256 * 4);
    float* HIDDEN = (float*)alloc((size_t)M * 1024 * 4);
    float* VALUE  = (float*)alloc((size_t)MV * 256 * 4);

    auto cvt = [&](const float* s, bf16* d, int n) {
        cvt_bf16_kernel<<<(n + 255) / 256, 256, 0, stream>>>(s, d, n);
    };
    cvt(sa_in_w,  W_SAIN,  768 * 256);
    cvt(sa_out_w, W_SAOUT, 256 * 256);
    cvt(off_w,    W_OFFAW,           128 * 256);
    cvt(aw_w,     W_OFFAW + 128*256, 128 * 256);
    cvt(val_w,    W_VAL,   256 * 256);
    cvt(proj_w,   W_PROJ,  256 * 256);
    cvt(lin1_w,   W_LIN1,  1024 * 256);
    cvt(lin2_w,   W_LIN2,  256 * 1024);
    copy_f32_kernel<<<1, 256, 0, stream>>>(off_b, B_OFFAW, 128);
    copy_f32_kernel<<<1, 256, 0, stream>>>(aw_b, B_OFFAW + 128, 128);

    const int BIG = 1 << 30;
    // 1) QKV: q/k inputs = tgt+pos (cols < 512), v input = tgt (cols >= 512)
    gemm_kernel<256><<<dim3(M / 16, 3), 128, 0, stream>>>(
        tgt, pos, tgt, nullptr, 512, W_SAIN, sa_in_b, nullptr, 0, QKV, 768);
    // 2) value projection (bandwidth-dominant GEMM)
    gemm_kernel<256><<<dim3(MV / 16, 1), 128, 0, stream>>>(
        memory, nullptr, memory, nullptr, BIG, W_VAL, val_b, nullptr, 0, VALUE, 256);
    // 3) pack attention operands
    pack_attn_kernel<<<32, 256, 0, stream>>>(QKV, QB, KB, VT);
    // 4) self-attention core
    attn_kernel<<<dim3(63, 32), 128, 0, stream>>>(QB, KB, VT, SA);
    // 5) out-proj + residual(tgt), then LN2 -> TGT2
    gemm_kernel<256><<<dim3(M / 16, 1), 128, 0, stream>>>(
        SA, nullptr, SA, nullptr, BIG, W_SAOUT, sa_out_b, tgt, 0, Y, 256);
    ln_kernel<<<M, 256, 0, stream>>>(Y, ln2_g, ln2_bb, TGT2);
    // 6) sampling offsets + attention weights from query = TGT2 + pos
    gemm_kernel<256><<<dim3(M / 16, 1), 128, 0, stream>>>(
        TGT2, pos, TGT2, pos, BIG, W_OFFAW, B_OFFAW, nullptr, 0, OFFAW, 256);
    offaw_post_kernel<<<M, 128, 0, stream>>>(OFFAW, ref, LOC, AWN);
    // 7) multi-scale deformable gather
    msdeform_kernel<<<M, 256, 0, stream>>>(VALUE, LOC, AWN, MSD);
    // 8) proj + residual(TGT2), LN1 -> TGT3
    gemm_kernel<256><<<dim3(M / 16, 1), 128, 0, stream>>>(
        MSD, nullptr, MSD, nullptr, BIG, W_PROJ, proj_b, TGT2, 0, Y, 256);
    ln_kernel<<<M, 256, 0, stream>>>(Y, ln1_g, ln1_bb, TGT3);
    // 9) FFN
    gemm_kernel<256><<<dim3(M / 16, 4), 128, 0, stream>>>(
        TGT3, nullptr, TGT3, nullptr, BIG, W_LIN1, lin1_b, nullptr, 1, HIDDEN, 1024);
    gemm_kernel<1024><<<dim3(M / 16, 1), 128, 0, stream>>>(
        HIDDEN, nullptr, HIDDEN, nullptr, BIG, W_LIN2, lin2_b, TGT3, 0, Y, 256);
    ln_kernel<<<M, 256, 0, stream>>>(Y, ln3_g, ln3_bb, (float*)d_out);
}
